// MLPQFunction_45449343926841
// MI455X (gfx1250) — compile-verified, matrix operand order checked by reading, not run on hardware
//
#include <hip/hip_runtime.h>

typedef _Float16 half_t;
typedef __attribute__((ext_vector_type(16))) _Float16 v16h;
typedef __attribute__((ext_vector_type(8)))  float    v8f;

#define B_ROWS   16384
#define PROWS    (16384 * 3)          // B * P
#define NSEG     14
#define XLD      (NSEG * 32)          // 448 halves per (b,p) row in Xall
#define ENC_LD   (NSEG * 64)          // 896 halves per (b,p) row in enc
#define LPAD     40                   // LDS row stride in halves (32 data + 8 pad)

#ifndef __has_builtin
#define __has_builtin(x) 0
#endif

#if __has_builtin(__builtin_amdgcn_global_load_async_to_lds_b128)
#define HAVE_ASYNC 1
#else
#define HAVE_ASYNC 0
#endif

#if __has_builtin(__builtin_amdgcn_s_wait_asynccnt)
#define WAIT_ASYNC(n) __builtin_amdgcn_s_wait_asynccnt(n)
#else
#define WAIT_ASYNC(n) asm volatile("s_wait_asynccnt %0" ::"i"(n) : "memory")
#endif

// async builtin takes (addrspace(1) int4* src, addrspace(3) int4* dst, imm, imm)
typedef int v4i __attribute__((vector_size(16)));
typedef __attribute__((address_space(1))) v4i gv4i;
typedef __attribute__((address_space(3))) v4i lv4i;

__device__ __forceinline__ float lrelu(float v) { return fmaxf(v, 0.01f * v); }

// Load one 16x32 f16 fragment (A/B-matrix layout per CDNA5 ISA 7.12.2):
//   lane L<16 : row M=L,    halves K=[k0..k0+7] then [k0+16..k0+23]
//   lane L>=16: row M=L-16, halves K=[k0+8..k0+15] then [k0+24..k0+31]
// p0 points at (row0, k0); ld in half elements. Works for LDS pointers too
// (inlined -> addrspace inferred -> ds_load_b128).
__device__ __forceinline__ v16h ld_frag(const half_t* p0, int ld) {
  const int lane = threadIdx.x & 31;
  const half_t* p = p0 + (size_t)(lane & 15) * ld + ((lane >> 4) << 3);
  union { v16h v; float4 q[2]; } u;
  u.q[0] = *(const float4*)(p);
  u.q[1] = *(const float4*)(p + 16);
  return u.v;
}

// Cooperatively stage a 128-row x 32-half tile (row stride ldg halves) into LDS
// rows padded to LPAD halves. 256 threads; each moves 2 x 16B.
// Async path: 4 ASYNCcnt ops per wave per call pair (2 for A + 2 for B).
__device__ __forceinline__ void stage_tile(const half_t* __restrict__ g0, int ldg,
                                           half_t* l0, int tid) {
#pragma unroll
  for (int i = 0; i < 2; ++i) {
    const int row = i * 64 + (tid >> 2);
    const int ch  = (tid & 3) * 8;   // halves
    const half_t* g = g0 + (size_t)row * ldg + ch;
    half_t* l = l0 + row * LPAD + ch;
#if HAVE_ASYNC
    __builtin_amdgcn_global_load_async_to_lds_b128((gv4i*)g, (lv4i*)l, 0, 0);
#else
    *(float4*)l = *(const float4*)g;
#endif
  }
}

// ---------------- weight packing ----------------

// W1T[s][n][k] (f16, k zero-padded to 32) = W1[s][k][n]
__global__ void k_pack_w1(const float* __restrict__ Wp1, const float* __restrict__ Wt1,
                          const float* __restrict__ Wr1, half_t* __restrict__ W1T) {
  int t = blockIdx.x * blockDim.x + threadIdx.x;
  if (t >= NSEG * 32 * 32) return;
  int s = t >> 10, n = (t >> 5) & 31, k = t & 31;
  float v = 0.0f;
  if (s < 9)       { if (k < 5)  v = Wp1[(s * 5 + k) * 32 + n]; }
  else if (s < 11) { if (k < 12) v = Wt1[((s - 9) * 12 + k) * 32 + n]; }
  else             { if (k < 9)  v = Wr1[((s - 11) * 9 + k) * 32 + n]; }
  W1T[t] = (half_t)v;
}

// W2T[s][n][k] (f16) = W2[s][k][n], K=32, N=64
__global__ void k_pack_w2(const float* __restrict__ Wp2, const float* __restrict__ Wt2,
                          const float* __restrict__ Wr2, half_t* __restrict__ W2T) {
  int t = blockIdx.x * blockDim.x + threadIdx.x;
  if (t >= NSEG * 64 * 32) return;
  int s = t >> 11, n = (t >> 5) & 63, k = t & 31;
  const float* src = (s < 9) ? (Wp2 + s * 2048)
                   : (s < 11) ? (Wt2 + (s - 9) * 2048)
                              : (Wr2 + (s - 11) * 2048);
  W2T[t] = (half_t)src[k * 64 + n];
}

// WT[n][k] (f16) = W[k][n] (f32). Mapped for coalesced f16 writes; strided f32
// reads are served out of the (L2-resident) source.
__global__ void k_transpose(const float* __restrict__ W, half_t* __restrict__ WT,
                            int K, int N) {
  int t = blockIdx.x * blockDim.x + threadIdx.x;
  if (t >= K * N) return;
  int n = t / K, k = t - n * K;
  WT[t] = (half_t)W[(size_t)k * N + n];
}

// ---------------- input gather: Xall[(b*3+p)][seg][0..31] f16 zero-padded ----------------
__global__ void k_gather(const float* __restrict__ obs, const float* __restrict__ act,
                         half_t* __restrict__ Xall) {
  int t = blockIdx.x * blockDim.x + threadIdx.x;
  if (t >= PROWS * NSEG) return;
  int row = t / NSEG, seg = t - row * NSEG;
  const float* o = obs + (size_t)row * 144;
  const float* a = act + (size_t)row * 3;
  union { half_t h[32]; float4 q[4]; } u;
#pragma unroll
  for (int i = 0; i < 32; ++i) u.h[i] = (half_t)0.0f;
  if (seg < 9) {
    u.h[0] = (half_t)o[2 * seg];
    u.h[1] = (half_t)o[2 * seg + 1];
    u.h[2] = (half_t)a[0]; u.h[3] = (half_t)a[1]; u.h[4] = (half_t)a[2];
  } else if (seg < 11) {
    int tt = seg - 9;
#pragma unroll
    for (int i = 0; i < 12; ++i) u.h[i] = (half_t)o[18 + 12 * tt + i];
  } else {
    int r = seg - 11;
#pragma unroll
    for (int i = 0; i < 9; ++i) u.h[i] = (half_t)o[117 + 9 * r + i];
  }
  float4* dst = (float4*)(Xall + (size_t)t * 32);
#pragma unroll
  for (int i = 0; i < 4; ++i) dst[i] = u.q[i];
}

// ---------------- per-segment encoder: 2-layer MLP via WMMA ----------------
// one wave handles one (16-row tile, segment) pair
__global__ __launch_bounds__(256) void k_encoder(
    const half_t* __restrict__ Xall, const half_t* __restrict__ W1T,
    const half_t* __restrict__ W2T,
    const float* __restrict__ bp1, const float* __restrict__ bt1, const float* __restrict__ br1,
    const float* __restrict__ bp2, const float* __restrict__ bt2, const float* __restrict__ br2,
    half_t* __restrict__ enc) {
  __shared__ __align__(16) half_t lds[8][16 * 32];
  const int wave = threadIdx.x >> 5;
  const int lane = threadIdx.x & 31;
  const int w    = blockIdx.x * 8 + wave;
  const int seg  = w % NSEG;
  const int row0 = (w / NSEG) * 16;

  const float* b1 = (seg < 9) ? bp1 + seg * 32
                  : (seg < 11) ? bt1 + (seg - 9) * 32 : br1 + (seg - 11) * 32;
  const float* b2 = (seg < 9) ? bp2 + seg * 64
                  : (seg < 11) ? bt2 + (seg - 9) * 64 : br2 + (seg - 11) * 64;

  const int cr = lane & 15;          // output column within 16-wide tile
  const int rb = (lane >> 4) << 3;   // row base 0 or 8

  // layer 1: [16x32] x [32 -> N=32]
  v16h a = ld_frag(Xall + (size_t)row0 * XLD + seg * 32, XLD);
  v8f zero = {};
  v8f acc1[2] = { zero, zero };
#pragma unroll
  for (int j = 0; j < 2; ++j) {
    v16h b = ld_frag(W1T + seg * 1024 + (j * 16) * 32, 32);
    acc1[j] = __builtin_amdgcn_wmma_f32_16x16x32_f16(false, a, false, b,
                                                     (short)0, acc1[j], false, false);
  }
#pragma unroll
  for (int j = 0; j < 2; ++j) {
    int col = j * 16 + cr;
    float bv = b1[col];
#pragma unroll
    for (int r = 0; r < 8; ++r)
      lds[wave][(rb + r) * 32 + col] = (half_t)lrelu(acc1[j][r] + bv);
  }
  __syncthreads();

  // layer 2: [16x32] x [32 -> N=64]
  v16h a2 = ld_frag(&lds[wave][0], 32);
  v8f acc2[4] = { zero, zero, zero, zero };
#pragma unroll
  for (int j = 0; j < 4; ++j) {
    v16h b = ld_frag(W2T + seg * 2048 + (j * 16) * 32, 32);
    acc2[j] = __builtin_amdgcn_wmma_f32_16x16x32_f16(false, a2, false, b,
                                                     (short)0, acc2[j], false, false);
  }
#pragma unroll
  for (int j = 0; j < 4; ++j) {
    int col = j * 16 + cr;
    float bv = b2[col];
#pragma unroll
    for (int r = 0; r < 8; ++r)
      enc[(size_t)(row0 + rb + r) * ENC_LD + seg * 64 + col] =
          (half_t)lrelu(acc2[j][r] + bv);
  }
}

// ---------------- main WMMA GEMM: C = relu(A[MxK] * BT[NxK]^T + bias) ----------------
// block = 8 waves (4M x 2N); block tile 128x128; wave tile 32x64; K-step 32.
// A/B K-slices staged through LDS with double-buffered async global->LDS copies.
__global__ __launch_bounds__(256) void k_gemm_bias_relu(
    const half_t* __restrict__ A, const half_t* __restrict__ BT,
    const float* __restrict__ bias, half_t* __restrict__ C,
    int M, int N, int K) {
  __shared__ __align__(16) half_t lA[2][128 * LPAD];
  __shared__ __align__(16) half_t lB[2][128 * LPAD];
  const int tid  = threadIdx.x;
  const int wave = tid >> 5;
  const int lane = tid & 31;
  const int rowB = blockIdx.y * 128;
  const int colB = blockIdx.x * 128;
  const int wr = (wave & 3) * 32;            // wave row offset in block tile
  const int wc = (wave >> 2) * 64;           // wave col offset in block tile
  (void)M;

  v8f zero = {};
  v8f acc[2][4];
#pragma unroll
  for (int i = 0; i < 2; ++i)
#pragma unroll
    for (int j = 0; j < 4; ++j) acc[i][j] = zero;

  // prologue: stage k0 = 0 into buffer 0 (4 async ops per wave)
  stage_tile(A + (size_t)rowB * K, K, &lA[0][0], tid);
  stage_tile(BT + (size_t)colB * K, K, &lB[0][0], tid);

  const int KT = K / 32;
  for (int kt = 0; kt < KT; ++kt) {
    const int buf = kt & 1;
    if (kt + 1 < KT) {           // stage next K-slice into the other buffer
      const int k1 = (kt + 1) * 32;
      stage_tile(A + (size_t)rowB * K + k1, K, &lA[buf ^ 1][0], tid);
      stage_tile(BT + (size_t)colB * K + k1, K, &lB[buf ^ 1][0], tid);
#if HAVE_ASYNC
      WAIT_ASYNC(4);             // current buffer's 4 ops done; next 4 in flight
#endif
    } else {
#if HAVE_ASYNC
      WAIT_ASYNC(0);
#endif
    }
    __syncthreads();             // current buffer visible to all waves

    v16h af[2], bf[4];
    af[0] = ld_frag(&lA[buf][(wr + 0)  * LPAD], LPAD);
    af[1] = ld_frag(&lA[buf][(wr + 16) * LPAD], LPAD);
#pragma unroll
    for (int j = 0; j < 4; ++j)
      bf[j] = ld_frag(&lB[buf][(wc + j * 16) * LPAD], LPAD);
#pragma unroll
    for (int i = 0; i < 2; ++i)
#pragma unroll
      for (int j = 0; j < 4; ++j)
        acc[i][j] = __builtin_amdgcn_wmma_f32_16x16x32_f16(
            false, af[i], false, bf[j], (short)0, acc[i][j], false, false);

    __syncthreads();             // buffer free before it is overwritten
  }

  const int cr = lane & 15;
  const int rb = (lane >> 4) << 3;
#pragma unroll
  for (int i = 0; i < 2; ++i)
#pragma unroll
    for (int j = 0; j < 4; ++j) {
      int col = colB + wc + j * 16 + cr;
      float bv = bias[col];
#pragma unroll
      for (int r = 0; r < 8; ++r) {
        int row = rowB + wr + i * 16 + rb + r;
        C[(size_t)row * N + col] = (half_t)fmaxf(acc[i][j][r] + bv, 0.0f);
      }
    }
}

// ---------------- final head: one wave per batch row, lane-parallel dot ----------------
__global__ __launch_bounds__(256) void k_head(const half_t* __restrict__ h2,
                                              const float* __restrict__ Wq3,
                                              const float* __restrict__ bq3,
                                              float* __restrict__ out) {
  const int wave = threadIdx.x >> 5;
  const int lane = threadIdx.x & 31;
  const int b = blockIdx.x * 8 + wave;
  const half_t* h = h2 + (size_t)b * 512 + lane * 16;
  union { v16h v; float4 q[2]; } u;
  u.q[0] = *(const float4*)(h);
  u.q[1] = *(const float4*)(h + 8);
  const float* w = Wq3 + (size_t)(lane * 16) * 3;
  float s0 = 0.f, s1 = 0.f, s2 = 0.f;
#pragma unroll
  for (int i = 0; i < 16; ++i) {
    float x = (float)u.v[i];
    s0 += x * w[i * 3 + 0];
    s1 += x * w[i * 3 + 1];
    s2 += x * w[i * 3 + 2];
  }
#pragma unroll
  for (int m = 16; m >= 1; m >>= 1) {
    s0 += __shfl_xor(s0, m, 32);
    s1 += __shfl_xor(s1, m, 32);
    s2 += __shfl_xor(s2, m, 32);
  }
  if (lane == 0) {
    out[b * 3 + 0] = s0 + bq3[0];
    out[b * 3 + 1] = s1 + bq3[1];
    out[b * 3 + 2] = s2 + bq3[2];
  }
}

extern "C" void kernel_launch(void* const* d_in, const int* in_sizes, int n_in,
                              void* d_out, int out_size, void* d_ws, size_t ws_size,
                              hipStream_t stream) {
  const float* obs = (const float*)d_in[0];
  const float* act = (const float*)d_in[1];
  const float* Wp1 = (const float*)d_in[2],  *bp1 = (const float*)d_in[3];
  const float* Wp2 = (const float*)d_in[4],  *bp2 = (const float*)d_in[5];
  const float* Wt1 = (const float*)d_in[6],  *bt1 = (const float*)d_in[7];
  const float* Wt2 = (const float*)d_in[8],  *bt2 = (const float*)d_in[9];
  const float* Wr1 = (const float*)d_in[10], *br1 = (const float*)d_in[11];
  const float* Wr2 = (const float*)d_in[12], *br2 = (const float*)d_in[13];
  const float* Wq1 = (const float*)d_in[14], *bq1 = (const float*)d_in[15];
  const float* Wq2 = (const float*)d_in[16], *bq2 = (const float*)d_in[17];
  const float* Wq3 = (const float*)d_in[18], *bq3 = (const float*)d_in[19];
  float* out = (float*)d_out;
  (void)in_sizes; (void)n_in; (void)out_size; (void)ws_size;

  char* ws = (char*)d_ws;
  size_t off = 0;
  auto alloc = [&](size_t bytes) {
    void* p = ws + off;
    off = (off + bytes + 255) & ~(size_t)255;
    return p;
  };
  half_t* Xall = (half_t*)alloc((size_t)PROWS * XLD * 2);     // 44 MB
  half_t* enc  = (half_t*)alloc((size_t)PROWS * ENC_LD * 2);  // 88 MB
  half_t* h1   = (half_t*)alloc((size_t)B_ROWS * 1024 * 2);   // 33.5 MB
  half_t* h2   = (half_t*)alloc((size_t)B_ROWS * 512 * 2);    // 16.8 MB
  half_t* Wq1T = (half_t*)alloc((size_t)1024 * 2688 * 2);     // 5.5 MB
  half_t* Wq2T = (half_t*)alloc((size_t)512 * 1024 * 2);      // 1 MB
  half_t* W1T  = (half_t*)alloc((size_t)NSEG * 32 * 32 * 2);
  half_t* W2T  = (half_t*)alloc((size_t)NSEG * 64 * 32 * 2);

  k_pack_w1<<<(NSEG * 32 * 32 + 255) / 256, 256, 0, stream>>>(Wp1, Wt1, Wr1, W1T);
  k_pack_w2<<<(NSEG * 64 * 32 + 255) / 256, 256, 0, stream>>>(Wp2, Wt2, Wr2, W2T);
  k_transpose<<<(2688 * 1024 + 255) / 256, 256, 0, stream>>>(Wq1, Wq1T, 2688, 1024);
  k_transpose<<<(1024 * 512 + 255) / 256, 256, 0, stream>>>(Wq2, Wq2T, 1024, 512);
  k_gather<<<(PROWS * NSEG + 255) / 256, 256, 0, stream>>>(obs, act, Xall);

  // (PROWS/16) row tiles * 14 segments = 43008 waves -> 5376 blocks of 8 waves
  k_encoder<<<(PROWS / 16) * NSEG / 8, 256, 0, stream>>>(
      Xall, W1T, W2T, bp1, bt1, br1, bp2, bt2, br2, enc);

  // q head: enc[B,2688] (== [PROWS,896] memory) -> h1[B,1024] -> h2[B,512]
  k_gemm_bias_relu<<<dim3(1024 / 128, B_ROWS / 128), 256, 0, stream>>>(
      enc, Wq1T, bq1, h1, B_ROWS, 1024, 2688);
  k_gemm_bias_relu<<<dim3(512 / 128, B_ROWS / 128), 256, 0, stream>>>(
      h1, Wq2T, bq2, h2, B_ROWS, 512, 1024);

  k_head<<<(B_ROWS * 3 + 255) / 256, 256, 0, stream>>>(h2, Wq3, bq3, out);
}
